// SparseMultiHeadAttention_65592740544536
// MI455X (gfx1250) — matrix operand order repaired and checked by reading, not compiled, not run on hardware
//
#include <hip/hip_runtime.h>
#include <hip/hip_bf16.h>

// ---------------------------------------------------------------------------
// Problem constants (fixed by the harness)
// ---------------------------------------------------------------------------
constexpr int Hh  = 1024;   // hidden dim
constexpr int Ss  = 4096;   // sequence length
constexpr int Bb  = 4;      // batch
constexpr int NSs = 2048;   // sampled positions (S/2)

// GEMM tiling
constexpr int BM = 64;      // block tile M
constexpr int BN = 256;     // block tile N
constexpr int BK = 64;      // K step (two WMMA K substeps)
constexpr int WM = 32;      // wave tile M (2 x 16)
constexpr int WN = 64;      // wave tile N (4 x 16)
constexpr int LDP = BK + 8; // padded LDS row (72 elems = 144 B, keeps 16B align)

typedef __bf16 bf16_t;
typedef bf16_t v4bf  __attribute__((ext_vector_type(4)));
typedef bf16_t v8bf  __attribute__((ext_vector_type(8)));
typedef bf16_t v16bf __attribute__((ext_vector_type(16)));
typedef float  v8f   __attribute__((ext_vector_type(8)));
typedef int    v4i   __attribute__((ext_vector_type(4)));

enum { EPI_BF16_ROW = 0, EPI_BF16_TRANS = 1, EPI_F32_SCALE = 2, EPI_F32_SCATTER = 3 };

// ---------------------------------------------------------------------------
// Async global->LDS path (CDNA5 GLOBAL_LOAD_ASYNC_TO_LDS_B128, ASYNCcnt).
// Falls back to the synchronous reg-staged path if the builtin is absent
// (e.g. on the host pass, where __has_builtin reports 0).
// ---------------------------------------------------------------------------
#if __has_builtin(__builtin_amdgcn_global_load_async_to_lds_b128)
#define USE_ASYNC 1
#else
#define USE_ASYNC 0
#endif

#if USE_ASYNC
#if __has_builtin(__builtin_amdgcn_s_wait_asynccnt)
#define ASYNC_WAIT(n) __builtin_amdgcn_s_wait_asynccnt(n)
#else
#define ASYNC_WAIT(n) asm volatile("s_wait_asynccnt %0" ::"i"(n) : "memory")
#endif
#define AS_GLOBAL __attribute__((address_space(1)))
#define AS_LDS    __attribute__((address_space(3)))
__device__ __forceinline__ void async_b128(const void* g, void* l) {
    __builtin_amdgcn_global_load_async_to_lds_b128(
        (AS_GLOBAL v4i*)g, (AS_LDS v4i*)l, 0, 0);
}
#else
#define ASYNC_WAIT(n) ((void)0)
#endif

// ---------------------------------------------------------------------------
// Generic C = A * B^T GEMM.  A: (M x K) row-major bf16, B: (N x K) row-major
// bf16, fp32 WMMA accumulation, double-buffered async LDS pipeline.
// grid = (N/BN, M/BM, nbatch); all dims divide evenly for this problem.
// ---------------------------------------------------------------------------
template <int EPI>
__global__ __launch_bounds__(256)
void gemm_bt(const bf16_t* __restrict__ A, int lda, long sA,
             const bf16_t* __restrict__ Bm, int ldb, long sB,
             void* __restrict__ C, int ldc, long sC,
             int K, float scale,
             const int* __restrict__ idx, const float* __restrict__ bias)
{
    __shared__ __align__(16) bf16_t As[2][BM][LDP];
    __shared__ __align__(16) bf16_t Bs[2][BN][LDP];

    const int tid  = threadIdx.x;
    const int lane = tid & 31;
    const int wave = tid >> 5;
    const int wm   = (wave >> 2) * WM;   // 0 / 32
    const int wn   = (wave & 3)  * WN;   // 0 / 64 / 128 / 192
    const long bm  = (long)blockIdx.y * BM;
    const long bn  = (long)blockIdx.x * BN;
    const long z   = blockIdx.z;

    const bf16_t* Ab = A  + z * sA;
    const bf16_t* Bp = Bm + z * sB;

    v8f acc[2][4] = {};

    const int lh   = lane & 15;
    const int aKlo = (lane < 16) ? 0 : 8;    // ISA 7.12.2: 16-bit A half-wave K split
    const int bKlo = (lane < 16) ? 0 : 8;    // B: same half-wave K split as A

    // Fill one LDS tile pair (A: 2 chunks/thread, B: 8 chunks/thread).
    auto issue_tile = [&](int k0, int buf) {
#if USE_ASYNC
#pragma unroll
        for (int i = 0; i < 2; ++i) {
            const int cid = tid + 256 * i;            // 0..511
            const int r = cid >> 3, kc = (cid & 7) * 8;
            async_b128(Ab + (bm + r) * (long)lda + k0 + kc, &As[buf][r][kc]);
        }
#pragma unroll
        for (int i = 0; i < 8; ++i) {
            const int cid = tid + 256 * i;            // 0..2047
            const int r = cid >> 3, kc = (cid & 7) * 8;
            async_b128(Bp + (bn + r) * (long)ldb + k0 + kc, &Bs[buf][r][kc]);
        }
#else
#pragma unroll
        for (int i = 0; i < 2; ++i) {
            const int cid = tid + 256 * i;
            const int r = cid >> 3, kc = (cid & 7) * 8;
            *reinterpret_cast<uint4*>(&As[buf][r][kc]) =
                *reinterpret_cast<const uint4*>(Ab + (bm + r) * (long)lda + k0 + kc);
        }
#pragma unroll
        for (int i = 0; i < 8; ++i) {
            const int cid = tid + 256 * i;
            const int r = cid >> 3, kc = (cid & 7) * 8;
            *reinterpret_cast<uint4*>(&Bs[buf][r][kc]) =
                *reinterpret_cast<const uint4*>(Bp + (bn + r) * (long)ldb + k0 + kc);
        }
#endif
    };

    auto compute_tile = [&](int buf) {
#pragma unroll
        for (int ks = 0; ks < BK; ks += 32) {
            v16bf af[2], bfv[4];
#pragma unroll
            for (int f = 0; f < 2; ++f) {
                v8bf lo = *reinterpret_cast<const v8bf*>(&As[buf][wm + f * 16 + lh][ks + aKlo]);
                v8bf hi = *reinterpret_cast<const v8bf*>(&As[buf][wm + f * 16 + lh][ks + aKlo + 16]);
                af[f] = __builtin_shufflevector(lo, hi, 0,1,2,3,4,5,6,7,8,9,10,11,12,13,14,15);
            }
#pragma unroll
            for (int g = 0; g < 4; ++g) {
                v8bf lo = *reinterpret_cast<const v8bf*>(&Bs[buf][wn + g * 16 + lh][ks + bKlo]);
                v8bf hi = *reinterpret_cast<const v8bf*>(&Bs[buf][wn + g * 16 + lh][ks + bKlo + 16]);
                bfv[g] = __builtin_shufflevector(lo, hi, 0,1,2,3,4,5,6,7,8,9,10,11,12,13,14,15);
            }
#pragma unroll
            for (int f = 0; f < 2; ++f)
#pragma unroll
                for (int g = 0; g < 4; ++g)
                    acc[f][g] = __builtin_amdgcn_wmma_f32_16x16x32_bf16(
                        false, af[f], false, bfv[g], (short)0, acc[f][g], false, false);
        }
    };

    // Double-buffered pipeline: tile k+1 streams into LDS while tile k computes.
    issue_tile(0, 0);
    int k0 = 0, buf = 0;
    for (; k0 + BK < K; k0 += BK, buf ^= 1) {
        __syncthreads();                 // everyone done reading buf^1
        issue_tile(k0 + BK, buf ^ 1);
        ASYNC_WAIT(10);                  // current tile's 10 ops done; next 10 in flight
        __syncthreads();                 // all waves' contributions to buf landed
        compute_tile(buf);
    }
    ASYNC_WAIT(0);
    __syncthreads();
    compute_tile(buf);

    // ---- epilogue ----
#pragma unroll
    for (int f = 0; f < 2; ++f) {
#pragma unroll
        for (int g = 0; g < 4; ++g) {
#pragma unroll
            for (int r = 0; r < 8; ++r) {
                const int  ml = wm + f * 16 + ((lane < 16) ? r : r + 8);
                const int  nl = wn + g * 16 + lh;
                const long gm = bm + ml;
                const long gn = bn + nl;
                float v = acc[f][g][r];
                if constexpr (EPI == EPI_BF16_ROW) {
                    if (bias) v += bias[gn];
                    ((bf16_t*)C)[z * sC + gm * ldc + gn] = (bf16_t)v;
                } else if constexpr (EPI == EPI_BF16_TRANS) {
                    if (bias) v += bias[gn];
                    ((bf16_t*)C)[z * sC + gn * ldc + gm] = (bf16_t)v;
                } else if constexpr (EPI == EPI_F32_SCALE) {
                    ((float*)C)[z * sC + gm * ldc + gn] = v * scale;
                } else { // EPI_F32_SCATTER: gm is flattened (batch*NS + q)
                    const long bidx = gm >> 11;          // / NS
                    const int  iq   = (int)(gm & (NSs - 1));
                    const long srow = idx[iq];
                    ((float*)C)[(bidx * Ss + srow) * Hh + gn] = v + bias[gn];
                }
            }
        }
    }
}

// ---------------------------------------------------------------------------
// Row softmax over 2048 fp32 scores; rewrites fp32 attn in place and emits a
// bf16 copy (P) for the attn @ V GEMM.  One block per (batch, query row).
// ---------------------------------------------------------------------------
__global__ __launch_bounds__(256)
void softmax_rows(float* __restrict__ attn, bf16_t* __restrict__ P)
{
    const long b   = blockIdx.y;
    const long row = blockIdx.x;
    float*  r = attn + (b * NSs + row) * (long)NSs;
    bf16_t* p = P    + (b * NSs + row) * (long)NSs;
    const int tid = threadIdx.x;

    float4 v0 = ((const float4*)r)[tid];
    float4 v1 = ((const float4*)r)[tid + 256];
    float f[8] = { v0.x, v0.y, v0.z, v0.w, v1.x, v1.y, v1.z, v1.w };

    __shared__ float redm[8];
    __shared__ float reds[8];

    float m = f[0];
#pragma unroll
    for (int i = 1; i < 8; ++i) m = fmaxf(m, f[i]);
#pragma unroll
    for (int o = 16; o > 0; o >>= 1) m = fmaxf(m, __shfl_xor(m, o, 32));
    if ((tid & 31) == 0) redm[tid >> 5] = m;
    __syncthreads();
    m = redm[0];
#pragma unroll
    for (int i = 1; i < 8; ++i) m = fmaxf(m, redm[i]);

    float s = 0.f;
#pragma unroll
    for (int i = 0; i < 8; ++i) { f[i] = __expf(f[i] - m); s += f[i]; }
#pragma unroll
    for (int o = 16; o > 0; o >>= 1) s += __shfl_xor(s, o, 32);
    if ((tid & 31) == 0) reds[tid >> 5] = s;
    __syncthreads();
    s = reds[0];
#pragma unroll
    for (int i = 1; i < 8; ++i) s += reds[i];
    const float inv = 1.0f / s;

    float4 o0 = { f[0] * inv, f[1] * inv, f[2] * inv, f[3] * inv };
    float4 o1 = { f[4] * inv, f[5] * inv, f[6] * inv, f[7] * inv };
    ((float4*)r)[tid]       = o0;
    ((float4*)r)[tid + 256] = o1;
    v4bf p0 = { (bf16_t)o0.x, (bf16_t)o0.y, (bf16_t)o0.z, (bf16_t)o0.w };
    v4bf p1 = { (bf16_t)o1.x, (bf16_t)o1.y, (bf16_t)o1.z, (bf16_t)o1.w };
    ((v4bf*)p)[tid]       = p0;
    ((v4bf*)p)[tid + 256] = p1;
}

// ---------------------------------------------------------------------------
// fp32 -> bf16 flat convert (weights). n must be multiple of 1024.
// ---------------------------------------------------------------------------
__global__ __launch_bounds__(256)
void convert_bf16(const float* __restrict__ src, bf16_t* __restrict__ dst)
{
    const long g = (long)blockIdx.x * 256 + threadIdx.x;
    float4 v = ((const float4*)src)[g];
    v4bf o = { (bf16_t)v.x, (bf16_t)v.y, (bf16_t)v.z, (bf16_t)v.w };
    ((v4bf*)dst)[g] = o;
}

// ---------------------------------------------------------------------------
// Gather sampled rows of x and convert to bf16.  One block per (b, i) row.
// ---------------------------------------------------------------------------
__global__ __launch_bounds__(256)
void gather_bf16(const float* __restrict__ x, const int* __restrict__ idx,
                 bf16_t* __restrict__ xs)
{
    const long r = blockIdx.x;          // 0 .. B*NS-1
    const long b = r >> 11;
    const int  i = (int)(r & (NSs - 1));
    const float* src = x + (b * Ss + idx[i]) * (long)Hh;
    bf16_t*      dst = xs + r * (long)Hh;
    const int t = threadIdx.x;          // 256 threads * float4 = 1024 elems
    float4 v = ((const float4*)src)[t];
    v4bf o = { (bf16_t)v.x, (bf16_t)v.y, (bf16_t)v.z, (bf16_t)v.w };
    ((v4bf*)dst)[t] = o;
}

// ---------------------------------------------------------------------------
// Initialize full y = broadcast(bo). Sampled rows get overwritten by the
// scatter GEMM epilogue.
// ---------------------------------------------------------------------------
__global__ __launch_bounds__(256)
void init_y(float* __restrict__ y, const float* __restrict__ bo)
{
    const long g  = (long)blockIdx.x * 256 + threadIdx.x;   // float4 index
    const int  n4 = (int)(g & (Hh / 4 - 1));
    ((float4*)y)[g] = ((const float4*)bo)[n4];
}

// ---------------------------------------------------------------------------
extern "C" void kernel_launch(void* const* d_in, const int* /*in_sizes*/, int /*n_in*/,
                              void* d_out, int /*out_size*/, void* d_ws, size_t /*ws_size*/,
                              hipStream_t stream)
{
    const float* x   = (const float*)d_in[0];
    const int*   idx = (const int*)  d_in[1];
    const float* Wq  = (const float*)d_in[2];
    const float* bq  = (const float*)d_in[3];
    const float* Wk  = (const float*)d_in[4];
    const float* bk  = (const float*)d_in[5];
    const float* Wv  = (const float*)d_in[6];
    const float* bv  = (const float*)d_in[7];
    const float* Wo  = (const float*)d_in[8];
    const float* bo  = (const float*)d_in[9];

    float* y    = (float*)d_out;
    float* attn = y + (long)Bb * Ss * Hh;

    // workspace layout (bytes); total 72 MB with lifetime-based reuse
    char* ws = (char*)d_ws;
    bf16_t* xs  = (bf16_t*)(ws);                    // 16 MB gathered x (bf16)
    bf16_t* wqb = (bf16_t*)(ws + (16l << 20));      //  2 MB
    bf16_t* wkb = (bf16_t*)(ws + (18l << 20));      //  2 MB
    bf16_t* wvb = (bf16_t*)(ws + (20l << 20));      //  2 MB
    bf16_t* wob = (bf16_t*)(ws + (22l << 20));      //  2 MB
    bf16_t* qs  = (bf16_t*)(ws + (24l << 20));      // 16 MB
    bf16_t* ksm = (bf16_t*)(ws + (40l << 20));      // 16 MB
    bf16_t* vst = (bf16_t*)(ws + (56l << 20));      // 16 MB  v^T per batch (H x NS)
    bf16_t* Pm   = qs;   // 32 MB, overlays qs+ks after scores are done
    bf16_t* outb = xs;   // 16 MB, overlays xs after projections are done

    const long M  = (long)Bb * NSs;      // 8192 flattened sampled rows
    const float scl = 1.0f / 32.0f;      // 1/sqrt(H)

    // 1) weights -> bf16
    convert_bf16<<<dim3(Hh * Hh / 1024), 256, 0, stream>>>(Wq, wqb);
    convert_bf16<<<dim3(Hh * Hh / 1024), 256, 0, stream>>>(Wk, wkb);
    convert_bf16<<<dim3(Hh * Hh / 1024), 256, 0, stream>>>(Wv, wvb);
    convert_bf16<<<dim3(Hh * Hh / 1024), 256, 0, stream>>>(Wo, wob);

    // 2) gather sampled x rows -> bf16
    gather_bf16<<<dim3((unsigned)M), 256, 0, stream>>>(x, idx, xs);

    // 3) y canvas = bias (non-sampled rows keep this value)
    init_y<<<dim3((unsigned)((long)Bb * Ss * Hh / 1024)), 256, 0, stream>>>(y, bo);

    // 4) q = xs @ Wq^T, k = xs @ Wk^T   (flattened M = 8192)
    gemm_bt<EPI_BF16_ROW><<<dim3(Hh / BN, (unsigned)(M / BM), 1), 256, 0, stream>>>(
        xs, Hh, 0, wqb, Hh, 0, qs, Hh, 0, Hh, 1.f, nullptr, bq);
    gemm_bt<EPI_BF16_ROW><<<dim3(Hh / BN, (unsigned)(M / BM), 1), 256, 0, stream>>>(
        xs, Hh, 0, wkb, Hh, 0, ksm, Hh, 0, Hh, 1.f, nullptr, bk);

    // 5) v^T[b] (H x NS) = transpose(xs[b] @ Wv^T)
    gemm_bt<EPI_BF16_TRANS><<<dim3(Hh / BN, NSs / BM, Bb), 256, 0, stream>>>(
        xs, Hh, (long)NSs * Hh, wvb, Hh, 0, vst, NSs, (long)Hh * NSs, Hh, 1.f, nullptr, bv);

    // 6) raw scores[b] = q[b] @ k[b]^T * scale  -> attn region of d_out (fp32)
    gemm_bt<EPI_F32_SCALE><<<dim3(NSs / BN, NSs / BM, Bb), 256, 0, stream>>>(
        qs, Hh, (long)NSs * Hh, ksm, Hh, (long)NSs * Hh,
        attn, NSs, (long)NSs * NSs, Hh, scl, nullptr, nullptr);

    // 7) softmax in place (fp32) + bf16 P copy
    softmax_rows<<<dim3(NSs, Bb), 256, 0, stream>>>(attn, Pm);

    // 8) out[b] = P[b] @ v[b]  (== P @ (v^T)^T), bf16 result
    gemm_bt<EPI_BF16_ROW><<<dim3(Hh / BN, NSs / BM, Bb), 256, 0, stream>>>(
        Pm, NSs, (long)NSs * NSs, vst, NSs, (long)Hh * NSs,
        outb, Hh, (long)NSs * Hh, NSs, 1.f, nullptr, nullptr);

    // 9) y rows at sampled indices = out @ Wo^T + bo (scatter epilogue)
    gemm_bt<EPI_F32_SCATTER><<<dim3(Hh / BN, (unsigned)(M / BM), 1), 256, 0, stream>>>(
        outb, Hh, 0, wob, Hh, 0, y, Hh, 0, Hh, 1.f, idx, bo);
}
